// SelfAttention_22170621182090
// MI455X (gfx1250) — matrix operand order repaired and checked by reading, not compiled, not run on hardware
//
#include <hip/hip_runtime.h>
#include <hip/hip_bf16.h>
#include <stdint.h>

// ---------------- CDNA5 WMMA types ----------------
typedef __attribute__((ext_vector_type(16))) __bf16 v16bf;
typedef __attribute__((ext_vector_type(8)))  float  v8f;

#define N_TOK 4096
#define DIM   1024
#define KPITCH 1032   // halves per LDS Keys row (1024 + 8 pad -> bank stagger, 16B aligned)
#define PPITCH 136    // halves per LDS P row   (128 + 8 pad)

__device__ __forceinline__ unsigned short f2bf(float f) {
    unsigned int u = __float_as_uint(f);
    unsigned int r = u + 0x7FFFu + ((u >> 16) & 1u);   // round-to-nearest-even
    return (unsigned short)(r >> 16);
}

__device__ __forceinline__ v16bf ld2q(const unsigned short* p0, const unsigned short* p1) {
    union { uint4 q[2]; v16bf v; } u;
    u.q[0] = *(const uint4*)p0;
    u.q[1] = *(const uint4*)p1;
    return u.v;
}

// A-matrix (16x32 bf16, MxK) from row-major bf16 matrix. Lane: M = lane&15,
// holds K = kb..kb+7 and kb+16..kb+23 with kb = 8*(lane>>4).  (ISA 7.12.2)
__device__ __forceinline__ v16bf load_A_g(const unsigned short* base, int row0, int pitch, int k0) {
    int lane = threadIdx.x & 31;
    const unsigned short* p = base + (size_t)(row0 + (lane & 15)) * pitch + k0 + ((lane >> 4) << 3);
    return ld2q(p, p + 16);
}
__device__ __forceinline__ v16bf load_A_lds(const unsigned short* sbase, int pitch, int k0) {
    int lane = threadIdx.x & 31;
    const unsigned short* p = sbase + (lane & 15) * pitch + k0 + ((lane >> 4) << 3);
    return ld2q(p, p + 16);
}
// B-matrix (32x16 bf16, KxN) where output column c is a contiguous row of `base`.
// Lane: N = lane&15, holds K = kb..kb+7 and kb+16..kb+23 with kb = 8*(lane>>4).
__device__ __forceinline__ v16bf load_B_g(const unsigned short* base, int col0, int pitch, int k0) {
    int lane = threadIdx.x & 31;
    const unsigned short* p = base + (size_t)(col0 + (lane & 15)) * pitch + k0 + ((lane >> 4) << 3);
    return ld2q(p, p + 16);
}

__device__ __forceinline__ v8f wmma_bf16(v16bf a, v16bf b, v8f c) {
    return __builtin_amdgcn_wmma_f32_16x16x32_bf16(false, a, false, b, (short)0, c, false, false);
}

// ---- CDNA5 async global->LDS copy (ASYNCcnt-tracked, bypasses VGPR data path) ----
__device__ __forceinline__ void async_g2l_b128(void* ldsp, const void* gp) {
    unsigned int  l = (unsigned int)(size_t)ldsp;           // LDS aperture: low 32 bits = LDS byte addr
    unsigned long long g = (unsigned long long)(size_t)gp;
    asm volatile("global_load_async_to_lds_b128 %0, %1, off" :: "v"(l), "v"(g) : "memory");
}
__device__ __forceinline__ void wait_async0() {
    asm volatile("s_wait_asynccnt 0x0" ::: "memory");
}

// reductions within a 16-lane half of the wave32 (C-matrix half)
__device__ __forceinline__ float hmax16(float v) {
    v = fmaxf(v, __shfl_xor(v, 1, 32));
    v = fmaxf(v, __shfl_xor(v, 2, 32));
    v = fmaxf(v, __shfl_xor(v, 4, 32));
    v = fmaxf(v, __shfl_xor(v, 8, 32));
    return v;
}
__device__ __forceinline__ float hsum16(float v) {
    v += __shfl_xor(v, 1, 32);
    v += __shfl_xor(v, 2, 32);
    v += __shfl_xor(v, 4, 32);
    v += __shfl_xor(v, 8, 32);
    return v;
}

// ---------------- Stage 0: conversions ----------------
__global__ void k_convert_x(const float4* __restrict__ X4, unsigned short* __restrict__ Xbf, int n4) {
    int i = blockIdx.x * blockDim.x + threadIdx.x;
    if (i >= n4) return;
    float4 f = X4[i];
    union { unsigned short h[4]; uint2 u; } p;
    p.h[0] = f2bf(f.x); p.h[1] = f2bf(f.y); p.h[2] = f2bf(f.z); p.h[3] = f2bf(f.w);
    *((uint2*)Xbf + i) = p.u;
}

// WT[n][k] = W[k][n] as bf16 (coalesced both ways via LDS tile)
__global__ void k_transpose_w(const float* __restrict__ W, unsigned short* __restrict__ WT) {
    __shared__ float tile[32][33];
    int k0 = blockIdx.y * 32, n0 = blockIdx.x * 32;
    int tx = threadIdx.x & 31, ty = threadIdx.x >> 5;   // 256 thr -> ty 0..7
    for (int r = ty; r < 32; r += 8)
        tile[r][tx] = W[(size_t)(k0 + r) * DIM + n0 + tx];
    __syncthreads();
    for (int r = ty; r < 32; r += 8)
        WT[(size_t)(n0 + r) * DIM + k0 + tx] = f2bf(tile[tx][r]);
}

// ---------------- Stage 1: projections  Y = X*W + b ----------------
// block = 128 thr (4 waves); each wave computes a 32x64 tile (2 A-tiles x 4 B-tiles,
// 8 WMMAs per 6 tile-loads). Ping-pong software pipeline, no rotation copies.
__global__ void __launch_bounds__(128) k_proj(const unsigned short* __restrict__ Xbf,
                                              const unsigned short* __restrict__ WT,
                                              const float* __restrict__ bias,
                                              unsigned short* __restrict__ out, int mode) {
    int wave = threadIdx.x >> 5, lane = threadIdx.x & 31;
    int m0 = blockIdx.y * 128 + wave * 32;
    int n0 = blockIdx.x * 64;
    v8f zero = {0.f, 0.f, 0.f, 0.f, 0.f, 0.f, 0.f, 0.f};
    v8f acc[8];                                  // [rt*4 + t]
#pragma unroll
    for (int i = 0; i < 8; ++i) acc[i] = zero;

    v16bf a0[2], b0[4];
    a0[0] = load_A_g(Xbf, m0,      DIM, 0);
    a0[1] = load_A_g(Xbf, m0 + 16, DIM, 0);
#pragma unroll
    for (int t = 0; t < 4; ++t) b0[t] = load_B_g(WT, n0 + t * 16, DIM, 0);

    for (int k = 0; k < DIM; k += 64) {
        int k1 = k + 32;                          // always < DIM (DIM % 64 == 0)
        v16bf a1[2], b1[4];
        a1[0] = load_A_g(Xbf, m0,      DIM, k1);
        a1[1] = load_A_g(Xbf, m0 + 16, DIM, k1);
#pragma unroll
        for (int t = 0; t < 4; ++t) b1[t] = load_B_g(WT, n0 + t * 16, DIM, k1);
#pragma unroll
        for (int t = 0; t < 4; ++t) {
            acc[t]     = wmma_bf16(a0[0], b0[t], acc[t]);
            acc[4 + t] = wmma_bf16(a0[1], b0[t], acc[4 + t]);
        }
        int k2 = (k + 64 < DIM) ? (k + 64) : k;   // harmless reload on last iter
        a0[0] = load_A_g(Xbf, m0,      DIM, k2);
        a0[1] = load_A_g(Xbf, m0 + 16, DIM, k2);
#pragma unroll
        for (int t = 0; t < 4; ++t) b0[t] = load_B_g(WT, n0 + t * 16, DIM, k2);
#pragma unroll
        for (int t = 0; t < 4; ++t) {
            acc[t]     = wmma_bf16(a1[0], b1[t], acc[t]);
            acc[4 + t] = wmma_bf16(a1[1], b1[t], acc[4 + t]);
        }
    }

    int h = lane >> 4, c = lane & 15;
#pragma unroll
    for (int rt = 0; rt < 2; ++rt) {
#pragma unroll
        for (int t = 0; t < 4; ++t) {
            int col = n0 + t * 16 + c;
            float bv = bias[col];
#pragma unroll
            for (int r = 0; r < 8; ++r) {
                int row = m0 + rt * 16 + h * 8 + r;
                unsigned short bf = f2bf(acc[rt * 4 + t][r] + bv);
                if (mode == 2) out[(size_t)col * N_TOK + row] = bf;   // VT[d][n]
                else           out[(size_t)row * DIM + col]   = bf;   // row-major
            }
        }
    }
}

// cooperative async stage of a 16-row Keys block into padded LDS
__device__ __forceinline__ void stage_keys(unsigned short* sK, const unsigned short* Kbf, int row0) {
    int tid = threadIdx.x;
    for (int idx = tid; idx < 16 * 128; idx += 256) {        // 128 x 16B per row
        int r = idx >> 7, q = idx & 127;
        async_g2l_b128(sK + r * KPITCH + q * 8, Kbf + (size_t)(row0 + r) * DIM + q * 8);
    }
    wait_async0();
    __syncthreads();
}

// score tile for columns [col0,col0+16): 4-buffer deep-pipelined K loop.
// Each Q tile is loaded ~3 WMMAs before its consumer; no rotation copies.
__device__ __forceinline__ v8f score_tile(const unsigned short* sK,
                                          const unsigned short* __restrict__ Qbf, int col0) {
    v8f s = {0.f, 0.f, 0.f, 0.f, 0.f, 0.f, 0.f, 0.f};
    v16bf q0 = load_B_g(Qbf, col0, DIM, 0);
    v16bf q1 = load_B_g(Qbf, col0, DIM, 32);
    v16bf q2 = load_B_g(Qbf, col0, DIM, 64);
    v16bf q3 = load_B_g(Qbf, col0, DIM, 96);
    for (int k = 0; k < DIM; k += 128) {
        int kp = (k + 128 < DIM) ? (k + 128) : k;   // harmless reloads on last iter
        s = wmma_bf16(load_A_lds(sK, KPITCH, k +  0), q0, s);
        q0 = load_B_g(Qbf, col0, DIM, kp +  0);
        s = wmma_bf16(load_A_lds(sK, KPITCH, k + 32), q1, s);
        q1 = load_B_g(Qbf, col0, DIM, kp + 32);
        s = wmma_bf16(load_A_lds(sK, KPITCH, k + 64), q2, s);
        q2 = load_B_g(Qbf, col0, DIM, kp + 64);
        s = wmma_bf16(load_A_lds(sK, KPITCH, k + 96), q3, s);
        q3 = load_B_g(Qbf, col0, DIM, kp + 96);
    }
    return s;
}

// ---------------- Stage 2: softmax stats (row max + sumexp) ----------------
__global__ void __launch_bounds__(256) k_attn_stats(const unsigned short* __restrict__ Kbf,
                                                    const unsigned short* __restrict__ Qbf,
                                                    float* __restrict__ gM, float* __restrict__ gL) {
    __shared__ __align__(16) unsigned short sK[16 * KPITCH];
    __shared__ float sm[8][16], sl[8][16];
    int tid = threadIdx.x, wave = tid >> 5, lane = tid & 31;
    int h = lane >> 4;
    int row0 = blockIdx.x * 16;
    stage_keys(sK, Kbf, row0);

    float mrun[8], lrun[8];
#pragma unroll
    for (int r = 0; r < 8; ++r) { mrun[r] = -1e30f; lrun[r] = 0.f; }

    for (int chunk = 0; chunk < 32; ++chunk) {
        v8f s = score_tile(sK, Qbf, (chunk * 8 + wave) * 16);
#pragma unroll
        for (int r = 0; r < 8; ++r) {
            float v = s[r];
            float tmax = hmax16(v);
            float tsum = hsum16(__expf(v - tmax));
            float mn = fmaxf(mrun[r], tmax);
            lrun[r] = lrun[r] * __expf(mrun[r] - mn) + tsum * __expf(tmax - mn);
            mrun[r] = mn;
        }
    }
    if ((lane & 15) == 0) {
#pragma unroll
        for (int r = 0; r < 8; ++r) { sm[wave][h * 8 + r] = mrun[r]; sl[wave][h * 8 + r] = lrun[r]; }
    }
    __syncthreads();
    if (tid < 16) {
        float m = -1e30f;
        for (int w = 0; w < 8; ++w) m = fmaxf(m, sm[w][tid]);
        float l = 0.f;
        for (int w = 0; w < 8; ++w) l += sl[w][tid] * __expf(sm[w][tid] - m);
        gM[row0 + tid] = m;
        gL[row0 + tid] = l;
    }
}

// ---------------- Stage 3: O = softmax(S) * V  + residual ----------------
__global__ void __launch_bounds__(256) k_attn_out(const unsigned short* __restrict__ Kbf,
                                                  const unsigned short* __restrict__ Qbf,
                                                  const unsigned short* __restrict__ VT,
                                                  const float* __restrict__ gM,
                                                  const float* __restrict__ gL,
                                                  const float* __restrict__ X,
                                                  float* __restrict__ XO) {
    __shared__ __align__(16) unsigned short sK[16 * KPITCH];
    __shared__ __align__(16) unsigned short sP[16 * PPITCH];
    int tid = threadIdx.x, wave = tid >> 5, lane = tid & 31;
    int h = lane >> 4, c = lane & 15;
    int row0 = blockIdx.x * 16;
    int d0 = wave * 128;

    stage_keys(sK, Kbf, row0);

    float mrow[8], linv[8];
#pragma unroll
    for (int r = 0; r < 8; ++r) {
        mrow[r] = gM[row0 + h * 8 + r];
        linv[r] = 1.f / gL[row0 + h * 8 + r];
    }
    v8f zero = {0.f, 0.f, 0.f, 0.f, 0.f, 0.f, 0.f, 0.f};
    v8f acc[8] = {zero, zero, zero, zero, zero, zero, zero, zero};

    for (int chunk = 0; chunk < 32; ++chunk) {
        v8f s = score_tile(sK, Qbf, (chunk * 8 + wave) * 16);
        __syncthreads();   // previous chunk's P fully consumed
#pragma unroll
        for (int r = 0; r < 8; ++r)
            sP[(h * 8 + r) * PPITCH + wave * 16 + c] = f2bf(__expf(s[r] - mrow[r]));
        __syncthreads();   // P chunk visible to all waves

        int kbase = chunk * 128;
        v16bf pa[4];
#pragma unroll
        for (int ss = 0; ss < 4; ++ss) pa[ss] = load_A_lds(sP, PPITCH, ss * 32);
        v16bf vb0 = load_B_g(VT, d0, N_TOK, kbase);
#pragma unroll
        for (int i = 0; i < 32; i += 2) {   // flattened (ss,t); fully unrolled
            int i1 = i + 1;
            v16bf vb1 = load_B_g(VT, d0 + (i1 & 7) * 16, N_TOK, kbase + (i1 >> 3) * 32);
            acc[i & 7] = wmma_bf16(pa[i >> 3], vb0, acc[i & 7]);
            int i2 = (i + 2 < 32) ? i + 2 : i;
            vb0 = load_B_g(VT, d0 + (i2 & 7) * 16, N_TOK, kbase + (i2 >> 3) * 32);
            acc[i1 & 7] = wmma_bf16(pa[i1 >> 3], vb1, acc[i1 & 7]);
        }
    }
    // epilogue: x = O/l + residual  (pre-LayerNorm)
#pragma unroll
    for (int t = 0; t < 8; ++t) {
#pragma unroll
        for (int r = 0; r < 8; ++r) {
            int row = row0 + h * 8 + r;
            int col = d0 + t * 16 + c;
            XO[(size_t)row * DIM + col] = acc[t][r] * linv[r] + X[(size_t)row * DIM + col];
        }
    }
}

// ---------------- Stage 4: LayerNorm ----------------
__global__ void __launch_bounds__(256) k_layernorm(const float* __restrict__ XO,
                                                   const float* __restrict__ gamma,
                                                   const float* __restrict__ beta,
                                                   float* __restrict__ out) {
    __shared__ float rs[8], rs2[8];
    int row = blockIdx.x, tid = threadIdx.x;
    int wave = tid >> 5, lane = tid & 31;
    const float* x = XO + (size_t)row * DIM;
    float v[4], s = 0.f, s2 = 0.f;
#pragma unroll
    for (int i = 0; i < 4; ++i) {
        v[i] = x[tid + i * 256];
        s += v[i];
        s2 += v[i] * v[i];
    }
    for (int m = 1; m < 32; m <<= 1) { s += __shfl_xor(s, m, 32); s2 += __shfl_xor(s2, m, 32); }
    if (lane == 0) { rs[wave] = s; rs2[wave] = s2; }
    __syncthreads();
    s = 0.f; s2 = 0.f;
    for (int w = 0; w < 8; ++w) { s += rs[w]; s2 += rs2[w]; }
    float mean = s * (1.f / DIM);
    float var = s2 * (1.f / DIM) - mean * mean;
    float rstd = rsqrtf(var + 1e-5f);
#pragma unroll
    for (int i = 0; i < 4; ++i) {
        int col = tid + i * 256;
        out[(size_t)row * DIM + col] = (v[i] - mean) * rstd * gamma[col] + beta[col];
    }
}

// ---------------- workspace layout ----------------
static const size_t OFF_XBF = 0;
static const size_t OFF_WKT = OFF_XBF + (size_t)N_TOK * DIM * 2;
static const size_t OFF_WQT = OFF_WKT + (size_t)DIM * DIM * 2;
static const size_t OFF_WVT = OFF_WQT + (size_t)DIM * DIM * 2;
static const size_t OFF_KBF = OFF_WVT + (size_t)DIM * DIM * 2;
static const size_t OFF_QBF = OFF_KBF + (size_t)N_TOK * DIM * 2;
static const size_t OFF_VT  = OFF_QBF + (size_t)N_TOK * DIM * 2;
static const size_t OFF_GM  = OFF_VT  + (size_t)DIM * N_TOK * 2;
static const size_t OFF_GL  = OFF_GM  + (size_t)N_TOK * 4;
static const size_t OFF_XO  = OFF_GL  + (size_t)N_TOK * 4;
static const size_t OFF_END = OFF_XO  + (size_t)N_TOK * DIM * 4;

extern "C" void kernel_launch(void* const* d_in, const int* in_sizes, int n_in,
                              void* d_out, int out_size, void* d_ws, size_t ws_size,
                              hipStream_t stream) {
    (void)in_sizes; (void)n_in; (void)out_size;
    if (ws_size < OFF_END) return;   // not enough scratch; nothing safe to do

    const float* X     = (const float*)d_in[0];
    const float* Wk    = (const float*)d_in[1];
    const float* bk    = (const float*)d_in[2];
    const float* Wq    = (const float*)d_in[3];
    const float* bq    = (const float*)d_in[4];
    const float* Wv    = (const float*)d_in[5];
    const float* bv    = (const float*)d_in[6];
    const float* gamma = (const float*)d_in[7];
    const float* beta  = (const float*)d_in[8];
    float* out = (float*)d_out;
    char* ws = (char*)d_ws;

    unsigned short* Xbf = (unsigned short*)(ws + OFF_XBF);
    unsigned short* WkT = (unsigned short*)(ws + OFF_WKT);
    unsigned short* WqT = (unsigned short*)(ws + OFF_WQT);
    unsigned short* WvT = (unsigned short*)(ws + OFF_WVT);
    unsigned short* Kbf = (unsigned short*)(ws + OFF_KBF);
    unsigned short* Qbf = (unsigned short*)(ws + OFF_QBF);
    unsigned short* VTb = (unsigned short*)(ws + OFF_VT);
    float* gM = (float*)(ws + OFF_GM);
    float* gL = (float*)(ws + OFF_GL);
    float* XO = (float*)(ws + OFF_XO);

    int n4 = N_TOK * DIM / 4;
    k_convert_x<<<n4 / 256, 256, 0, stream>>>((const float4*)X, Xbf, n4);
    dim3 tg(DIM / 32, DIM / 32);
    k_transpose_w<<<tg, 256, 0, stream>>>(Wk, WkT);
    k_transpose_w<<<tg, 256, 0, stream>>>(Wq, WqT);
    k_transpose_w<<<tg, 256, 0, stream>>>(Wv, WvT);

    dim3 pg(DIM / 64, N_TOK / 128);
    k_proj<<<pg, 128, 0, stream>>>(Xbf, WkT, bk, Kbf, 0);
    k_proj<<<pg, 128, 0, stream>>>(Xbf, WqT, bq, Qbf, 1);
    k_proj<<<pg, 128, 0, stream>>>(Xbf, WvT, bv, VTb, 2);

    k_attn_stats<<<N_TOK / 16, 256, 0, stream>>>(Kbf, Qbf, gM, gL);
    k_attn_out<<<N_TOK / 16, 256, 0, stream>>>(Kbf, Qbf, VTb, gM, gL, X, XO);
    k_layernorm<<<N_TOK, 256, 0, stream>>>(XO, gamma, beta, out);
}
